// PixelContrastLossOnlyNeg_22436909154963
// MI455X (gfx1250) — matrix operand order repaired and checked by reading, not compiled, not run on hardware
//
#include <hip/hip_runtime.h>
#include <hip/hip_bf16.h>

typedef __attribute__((ext_vector_type(16))) _Float16 v16h;
typedef __attribute__((ext_vector_type(8)))  float    v8f;
typedef __attribute__((ext_vector_type(4)))  float    v4f;   // native vector for nontemporal builtins
typedef __attribute__((ext_vector_type(4)))  unsigned int v4u;

#define C_DIM     256          // feature channels
#define M_CLS     256          // queue classes
#define HW        16384        // h*w = 128*128
#define TILE_PIX  64           // pixels per block
#define PITCH     264          // f16 per pixel row in LDS (256 + 8 pad -> bank-conflict-free)
#define NWAVE_TOT 16384        // 4096 blocks * 4 waves
#define INV_T     14.285714285714285714f  // 1/0.07

// ---- one-shot queue fp32 -> f16 conversion (L2-resident, 256KB -> 128KB) ----
__global__ __launch_bounds__(256)
void pcl_qcvt(const float* __restrict__ q, _Float16* __restrict__ qh)
{
    const int i = blockIdx.x * 256 + threadIdx.x;   // 16384 quads
    v4f v = *(const v4f*)(q + (size_t)i * 4);
    union { _Float16 h[4]; unsigned long long u; } o;
    o.h[0] = (_Float16)v.x; o.h[1] = (_Float16)v.y;
    o.h[2] = (_Float16)v.z; o.h[3] = (_Float16)v.w;
    *(unsigned long long*)(qh + (size_t)i * 4) = o.u;
}

__global__ __launch_bounds__(128)
void pcl_main(const float* __restrict__ feats,
              const _Float16* __restrict__ qh,
              const int*   __restrict__ labels,
              float* __restrict__ wsum,
              float* __restrict__ wcnt)
{
    __shared__ _Float16 ldsF[TILE_PIX * PITCH];

    const int tid   = threadIdx.x;
    const int nBase = blockIdx.x * TILE_PIX;   // global pixel-row base
    const int b_i   = nBase >> 14;             // / 16384
    const int p0    = nBase & (HW - 1);

    // ---- stage feats tile into LDS as f16, [pix][c] with padded pitch ----
    // thread t: pixel-group = t%16 (4 pixels), channel = t/16 + 8*iter
    {
        const int pg = tid & 15;
        const int c0 = tid >> 4;
        const float* fb = feats + ((size_t)(b_i * C_DIM) << 14) + p0 + pg * 4;
#pragma unroll 4
        for (int it = 0; it < 32; ++it) {
            const int c = c0 + it * 8;
            v4f v = __builtin_nontemporal_load((const v4f*)(fb + ((size_t)c << 14)));
            _Float16* dst = &ldsF[(pg * 4) * PITCH + c];
            dst[0 * PITCH] = (_Float16)v.x;
            dst[1 * PITCH] = (_Float16)v.y;
            dst[2 * PITCH] = (_Float16)v.z;
            dst[3 * PITCH] = (_Float16)v.w;
        }
    }
    __syncthreads();

    const int lane = tid & 31;
    const int wv   = tid >> 5;       // wave in block (0..3) -> 16-pixel slab
    const int hf   = lane >> 4;      // half-wave id
    const int l16  = lane & 15;

    // A-matrix row this lane feeds (ISA 16-bit A 16x32 layout)
    const _Float16* aRow = &ldsF[(wv * 16 + l16) * PITCH];
    // C/D layout: lane = column group, VGPR r = row; rows r and r+8 split by half
    const int rowBase = nBase + wv * 16 + hf * 8;
    const int* labBase = labels + (size_t)rowBase * M_CLS;

    float sumNeg[8], sumPos[8];
#pragma unroll
    for (int r = 0; r < 8; ++r) { sumNeg[r] = 0.f; sumPos[r] = 0.f; }

    for (int ct = 0; ct < 16; ++ct) {
        const int cls = ct * 16 + l16;               // this lane's class column
        const _Float16* qRow = qh + (size_t)cls * C_DIM;

        // issue label loads early; latency hidden under the WMMA chain
        int labs[8];
#pragma unroll
        for (int r = 0; r < 8; ++r)
            labs[r] = __builtin_nontemporal_load(labBase + (size_t)r * M_CLS + cls);

        v8f acc = {};
#pragma unroll
        for (int kc = 0; kc < 8; ++kc) {
            const int k0 = kc * 32;
            // A fragment: lanes 0-15 hold K 0-7 & 16-23; lanes 16-31 hold K 8-15 & 24-31
            union { v4u q[2]; v16h h; } A;
            A.q[0] = *(const v4u*)(aRow + k0 + hf * 8);
            A.q[1] = *(const v4u*)(aRow + k0 + 16 + hf * 8);
            // B fragment: lanes 0-15 hold K 0-7 & 16-23; lanes 16-31 hold K 8-15 & 24-31
            union { v4u q[2]; v16h h; } B;
            B.q[0] = *(const v4u*)(qRow + k0 + hf * 8);
            B.q[1] = *(const v4u*)(qRow + k0 + 16 + hf * 8);
            acc = __builtin_amdgcn_wmma_f32_16x16x32_f16(
                /*neg_a=*/false, A.h, /*neg_b=*/false, B.h,
                /*c_mod=*/(short)0, acc, /*reuse_a=*/false, /*reuse_b=*/false);
        }

        // branchless epilogue on the 16x16 logits tile: exp(+/-) with cndmask select
#pragma unroll
        for (int r = 0; r < 8; ++r) {
            const float lg = acc[r] * INV_T;
            const float ep = __expf(lg);
            const float en = __expf(-lg);
            const bool  p  = labs[r] != 0;
            sumPos[r] += p ? en : 0.f;
            sumNeg[r] += p ? 0.f : ep;
        }
    }

    // reduce across the 16 class-columns of each half-wave (xor masks stay in-half)
#pragma unroll
    for (int r = 0; r < 8; ++r) {
        float sn = sumNeg[r], sp = sumPos[r];
#pragma unroll
        for (int m = 1; m <= 8; m <<= 1) {
            sn += __shfl_xor(sn, m, 32);
            sp += __shfl_xor(sp, m, 32);
        }
        sumNeg[r] = sn; sumPos[r] = sp;
    }

    float lsum = 0.f, lcnt = 0.f;
    if (l16 == 0) {
#pragma unroll
        for (int r = 0; r < 8; ++r) {
            const float x = sumNeg[r] * sumPos[r];
            const float loss = __logf(x + 1.0f);
            lsum += loss;
            lcnt += (x > 0.f) ? 1.f : 0.f;
        }
    }
    lsum += __shfl_down(lsum, 16, 32);
    lcnt += __shfl_down(lcnt, 16, 32);
    if (lane == 0) {
        const int wid = blockIdx.x * 4 + wv;
        wsum[wid] = lsum;
        wcnt[wid] = lcnt;
    }
}

__global__ __launch_bounds__(256)
void pcl_reduce(const float* __restrict__ wsum,
                const float* __restrict__ wcnt,
                float* __restrict__ out)
{
    __shared__ float sS[256];
    __shared__ float sC[256];
    const int t = threadIdx.x;
    float s = 0.f, c = 0.f;
    for (int i = t; i < NWAVE_TOT; i += 256) { s += wsum[i]; c += wcnt[i]; }
    sS[t] = s; sC[t] = c;
    __syncthreads();
    for (int st = 128; st > 0; st >>= 1) {
        if (t < st) { sS[t] += sS[t + st]; sC[t] += sC[t + st]; }
        __syncthreads();
    }
    if (t == 0) {
        const float cnt = sC[0];
        out[0] = (cnt == 0.f) ? 0.f : sS[0] / cnt;
    }
}

extern "C" void kernel_launch(void* const* d_in, const int* in_sizes, int n_in,
                              void* d_out, int out_size, void* d_ws, size_t ws_size,
                              hipStream_t stream)
{
    const float* feats  = (const float*)d_in[0];  // [16,256,128,128] f32
    const float* queue  = (const float*)d_in[1];  // [256,256] f32
    const int*   labels = (const int*)d_in[2];    // [262144,256] i32

    float*    wsum = (float*)d_ws;                    // 16384 floats
    float*    wcnt = wsum + NWAVE_TOT;                // 16384 floats
    _Float16* qh   = (_Float16*)(wcnt + NWAVE_TOT);   // 65536 f16 (128KB)

    pcl_qcvt<<<64, 256, 0, stream>>>(queue, qh);
    pcl_main<<<4096, 128, 0, stream>>>(feats, qh, labels, wsum, wcnt);
    pcl_reduce<<<1, 256, 0, stream>>>(wsum, wcnt, (float*)d_out);
}
